// lstm_crf_21646635172305
// MI455X (gfx1250) — matrix operand order repaired and checked by reading, not compiled, not run on hardware
//
#include <hip/hip_runtime.h>
#include <hip/hip_bf16.h>

// BiLSTM-CRF NLL for MI455X (gfx1250, wave32, WMMA).
// Sizes fixed per the reference: B=128, S=512, V=50000, E=300 (pad 320), H=128, T=12.

typedef __attribute__((ext_vector_type(16))) _Float16 v16h;
typedef __attribute__((ext_vector_type(8)))  _Float16 v8h;
typedef __attribute__((ext_vector_type(8)))  float    v8f;

#define BS   65536   // B*S
#define SEQ  512
#define BAT  128
#define EPAD 320
#define G4H  512     // 4*H
#define HDIM 128
#define TTAG 12
#define SOS_ 3
#define EOS_ 4
#define NEGV (-10000.0f)

// ---------------- WMMA fragment helpers (contiguous k order, same for A and B) ----------------
// A fragment 16x32 f16, row-major source with leading dim `lda` (f16 elems).
// lane<16: K 0..15 ; lane>=16: K 16..31 ; M = lane&15.
__device__ __forceinline__ v16h load_a_frag(const _Float16* A, int lda, int lane) {
  const int m  = lane & 15;
  const int kb = (lane < 16) ? 0 : 16;
  const v8h lo = *(const v8h*)(A + m * lda + kb);
  const v8h hi = *(const v8h*)(A + m * lda + kb + 8);
  v16h r;
#pragma unroll
  for (int i = 0; i < 8; ++i) { r[i] = lo[i]; r[8 + i] = hi[i]; }
  return r;
}
// B fragment 32x16: source is W row-major [N][K] (i.e. B transposed), leading dim ldk.
// lane<16: K 0..15 ; lane>=16: K 16..31 ; N = lane&15.  One contiguous 32B load.
__device__ __forceinline__ v16h load_b_frag(const _Float16* W, int ldk, int lane) {
  const int n  = lane & 15;
  const int kb = (lane < 16) ? 0 : 16;
  return *(const v16h*)(W + n * ldk + kb);
}

__device__ __forceinline__ float sigm(float x) { return 1.0f / (1.0f + __expf(-x)); }

// ---------------- 1) embedding gather + f32->f16, pad E 300->320 ----------------
__global__ void k_gather(const int* __restrict__ x, const float* __restrict__ embed,
                         _Float16* __restrict__ emb16) {
  const int i = blockIdx.x;          // 0..BS-1
  const int k = threadIdx.x;         // 0..319
  const int tok = x[i];
  const float v = (k < 300) ? embed[(size_t)tok * 300 + k] : 0.0f;
  emb16[(size_t)i * EPAD + k] = (_Float16)v;
}

// ---------------- 2) generic weight convert f32[R,C] -> f16[Rout,Cout] (zero pad) ----
__global__ void k_convert(const float* __restrict__ src, _Float16* __restrict__ dst,
                          int R, int C, int Rout, int Cout) {
  const int idx = blockIdx.x * blockDim.x + threadIdx.x;
  if (idx >= Rout * Cout) return;
  const int r = idx / Cout, c = idx % Cout;
  dst[idx] = (r < R && c < C) ? (_Float16)src[r * C + c] : (_Float16)0.0f;
}

// ---------------- 3) input projection GEMM: G[BS,512] = emb16 @ Wih16^T + b ---------
// grid (4096, 4), block 256 (8 waves); wave -> one 16x16 tile, 10 K-tiles.
__global__ void k_inproj(const _Float16* __restrict__ A,     // [BS, 320]
                         const _Float16* __restrict__ W,     // [512, 320]
                         const float* __restrict__ bias,     // [512]
                         float* __restrict__ G) {            // [BS, 512]
  const int lane  = threadIdx.x & 31;
  const int wv    = threadIdx.x >> 5;
  const int mtile = blockIdx.x;
  const int ntile = blockIdx.y * 8 + wv;
  const _Float16* Abase = A + (size_t)mtile * 16 * EPAD;
  const _Float16* Wbase = W + (size_t)ntile * 16 * EPAD;
  v8f acc = {};
#pragma unroll
  for (int kt = 0; kt < 10; ++kt) {
    v16h a = load_a_frag(Abase + kt * 32, EPAD, lane);
    v16h b = load_b_frag(Wbase + kt * 32, EPAD, lane);
    acc = __builtin_amdgcn_wmma_f32_16x16x32_f16(false, a, false, b, (short)0, acc,
                                                 false, false);
  }
  const int   n     = ntile * 16 + (lane & 15);
  const float bn    = bias[n];
  const int   mrow0 = mtile * 16 + ((lane >> 4) << 3);
#pragma unroll
  for (int r = 0; r < 8; ++r)
    G[(size_t)(mrow0 + r) * G4H + n] = acc[r] + bn;
}

// ---------------- 4) recurrent LSTM, both directions in one launch -----------------
// grid 16: blocks 0..7 forward (batch tiles), 8..15 backward. block 256 (8 waves).
// LDS: Whh16 [512,128] f16 (128KB) + h ping-pong 2x[16,128] f16 (8KB).
// Wave w owns h-channels j = 16w..16w+15 across all four gates (tiles w, w+8, w+16, w+24).
__global__ void k_lstm_rec(const _Float16* __restrict__ WhhF,  // [512,128] f16
                           const _Float16* __restrict__ WhhB,
                           const float* __restrict__ Gf,       // [BS,512]
                           const float* __restrict__ Gb,
                           _Float16* __restrict__ hcat) {      // [BS,256] f16
  extern __shared__ char smem[];
  _Float16* sWhh = (_Float16*)smem;                       // 512*128
  _Float16* sH0  = (_Float16*)(smem + G4H * HDIM * 2);    // 16*128
  _Float16* sH1  = sH0 + 16 * HDIM;

  const int dir = blockIdx.x >> 3;
  const int b0  = (blockIdx.x & 7) * 16;
  const _Float16* Whh = dir ? WhhB : WhhF;
  const float*    G   = dir ? Gb : Gf;
  const int col_off   = dir * HDIM;

  const int tid  = threadIdx.x;
  const int lane = tid & 31;
  const int wv   = tid >> 5;

  // cooperative load Whh into LDS (8 f16 per iter per thread) + zero ping buffer
  {
    const v8h* src = (const v8h*)Whh;
    v8h*       dst = (v8h*)sWhh;
    for (int i = tid; i < (G4H * HDIM) / 8; i += 256) dst[i] = src[i];
    for (int i = tid; i < 16 * HDIM; i += 256) sH0[i] = (_Float16)0.0f;
  }
  __syncthreads();

  // hoist loop-invariant B fragments: [gate 0..3][ktile 0..3]
  v16h bfrag[4][4];
#pragma unroll
  for (int g = 0; g < 4; ++g)
#pragma unroll
    for (int kt = 0; kt < 4; ++kt)
      bfrag[g][kt] = load_b_frag(sWhh + (size_t)(g * 8 + wv) * 16 * HDIM + kt * 32,
                                 HDIM, lane);

  const int j     = 16 * wv + (lane & 15);   // h channel owned by this thread
  const int mrow  = (lane >> 4) << 3;        // 0 or 8
  float c_state[8];
#pragma unroll
  for (int r = 0; r < 8; ++r) c_state[r] = 0.0f;

  _Float16* hb[2] = {sH0, sH1};
  int p = 0;
  for (int t = 0; t < SEQ; ++t) {
    const int ts = dir ? (SEQ - 1 - t) : t;
    // prefetch next step's gate rows (global_prefetch_b8)
    if (t + 1 < SEQ) {
      const int tsn = dir ? (SEQ - 2 - t) : (t + 1);
      const size_t rn = ((size_t)(b0 + mrow) * SEQ + tsn) * G4H + j;
      __builtin_prefetch(G + rn, 0, 1);
      __builtin_prefetch(G + rn + 256, 0, 1);
    }

    v16h afrag[4];
#pragma unroll
    for (int kt = 0; kt < 4; ++kt)
      afrag[kt] = load_a_frag(hb[p] + kt * 32, HDIM, lane);

    v8f acc[4] = {v8f{}, v8f{}, v8f{}, v8f{}};
#pragma unroll
    for (int g = 0; g < 4; ++g)
#pragma unroll
      for (int kt = 0; kt < 4; ++kt)
        acc[g] = __builtin_amdgcn_wmma_f32_16x16x32_f16(false, afrag[kt], false,
                                                        bfrag[g][kt], (short)0,
                                                        acc[g], false, false);
#pragma unroll
    for (int r = 0; r < 8; ++r) {
      const int m = mrow + r;
      const size_t row = (size_t)(b0 + m) * SEQ + ts;
      const float gi = acc[0][r] + G[row * G4H + j];
      const float gf = acc[1][r] + G[row * G4H + HDIM + j];
      const float gg = acc[2][r] + G[row * G4H + 2 * HDIM + j];
      const float go = acc[3][r] + G[row * G4H + 3 * HDIM + j];
      const float c  = sigm(gf) * c_state[r] + sigm(gi) * tanhf(gg);
      c_state[r] = c;
      const float h = sigm(go) * tanhf(c);
      const _Float16 h16 = (_Float16)h;
      hcat[row * 256 + col_off + j] = h16;   // for the output projection
      hb[p ^ 1][m * HDIM + j] = h16;         // next step's A matrix
    }
    __syncthreads();
    p ^= 1;
  }
}

// ---------------- 5) emission GEMM: Y[BS,16] = hcat @ Wout16^T + bout ---------------
// grid 512, block 256; wave -> one M tile, single N tile (T=12 padded to 16), K=8 tiles.
__global__ void k_emit(const _Float16* __restrict__ hcat,   // [BS,256]
                       const _Float16* __restrict__ Wout16, // [16,256] (rows 12..15 zero)
                       const float* __restrict__ bout,      // [12]
                       float* __restrict__ Y) {              // [BS,16]
  const int lane  = threadIdx.x & 31;
  const int wv    = threadIdx.x >> 5;
  const int mtile = blockIdx.x * 8 + wv;
  const _Float16* Abase = hcat + (size_t)mtile * 16 * 256;
  v8f acc = {};
#pragma unroll
  for (int kt = 0; kt < 8; ++kt) {
    v16h a = load_a_frag(Abase + kt * 32, 256, lane);
    v16h b = load_b_frag(Wout16 + kt * 32, 256, lane);
    acc = __builtin_amdgcn_wmma_f32_16x16x32_f16(false, a, false, b, (short)0, acc,
                                                 false, false);
  }
  const int   n     = lane & 15;
  const float bn    = (n < TTAG) ? bout[n] : 0.0f;
  const int   mrow0 = mtile * 16 + ((lane >> 4) << 3);
#pragma unroll
  for (int r = 0; r < 8; ++r)
    Y[(size_t)(mrow0 + r) * 16 + n] = acc[r] + bn;
}

// ---------------- 6) CRF forward recursion + gold score -----------------------------
// grid 8, block 256: 16 batches x 16 lanes (12 active states).
__global__ void k_crf(const float* __restrict__ Y,     // [BS,16]
                      const int* __restrict__ y0,      // [B,S]
                      const float* __restrict__ trans, // [12,12]
                      float* __restrict__ out) {       // [B]
  __shared__ float sTr[TTAG * TTAG];
  __shared__ float sc[16][TTAG];
  const int tid = threadIdx.x;
  if (tid < TTAG * TTAG) sTr[tid] = trans[tid];
  const int bb = tid >> 4;
  const int i  = tid & 15;
  const int b  = blockIdx.x * 16 + bb;
  if (i < TTAG) sc[bb][i] = (i == SOS_) ? 0.0f : NEGV;
  __syncthreads();

  for (int s = 0; s < SEQ; ++s) {
    float nw = NEGV;
    if (i < TTAG) {
      float m = -3.0e38f;
#pragma unroll
      for (int jj = 0; jj < TTAG; ++jj) {
        const float v = sc[bb][jj] + sTr[i * TTAG + jj];
        m = fmaxf(m, v);
      }
      float sum = 0.0f;
#pragma unroll
      for (int jj = 0; jj < TTAG; ++jj)
        sum += __expf(sc[bb][jj] + sTr[i * TTAG + jj] - m);
      nw = m + __logf(sum) + Y[((size_t)b * SEQ + s) * 16 + i];
    }
    __syncthreads();
    if (i < TTAG) sc[bb][i] = nw;
    __syncthreads();
  }

  if (i == 0) {
    float m = -3.0e38f;
#pragma unroll
    for (int jj = 0; jj < TTAG; ++jj) m = fmaxf(m, sc[bb][jj] + sTr[EOS_ * TTAG + jj]);
    float sum = 0.0f;
#pragma unroll
    for (int jj = 0; jj < TTAG; ++jj) sum += __expf(sc[bb][jj] + sTr[EOS_ * TTAG + jj] - m);
    const float Z = m + __logf(sum);

    int prev = SOS_;
    float gold = 0.0f;
    for (int s = 0; s < SEQ; ++s) {
      const int t = y0[b * SEQ + s];
      gold += Y[((size_t)b * SEQ + s) * 16 + t] + sTr[t * TTAG + prev];
      prev = t;
    }
    gold += sTr[EOS_ * TTAG + prev];
    out[b] = Z - gold;
  }
}

// ---------------- launcher ----------------------------------------------------------
extern "C" void kernel_launch(void* const* d_in, const int* in_sizes, int n_in,
                              void* d_out, int out_size, void* d_ws, size_t ws_size,
                              hipStream_t stream) {
  const int*   x     = (const int*)d_in[0];
  const int*   y0    = (const int*)d_in[1];
  const float* embed = (const float*)d_in[2];
  const float* Wih_f = (const float*)d_in[3];
  const float* Whh_f = (const float*)d_in[4];
  const float* b_f   = (const float*)d_in[5];
  const float* Wih_b = (const float*)d_in[6];
  const float* Whh_b = (const float*)d_in[7];
  const float* b_b   = (const float*)d_in[8];
  const float* Wout  = (const float*)d_in[9];
  const float* bout  = (const float*)d_in[10];
  const float* trans = (const float*)d_in[11];
  float* out = (float*)d_out;

  // workspace carve (~350 MB assumed available)
  char* p = (char*)d_ws;
  auto take = [&](size_t bytes) { char* r = p; p += (bytes + 255) & ~(size_t)255; return r; };
  _Float16* emb16   = (_Float16*)take((size_t)BS * EPAD * 2);
  _Float16* Wih16f  = (_Float16*)take((size_t)G4H * EPAD * 2);
  _Float16* Wih16b  = (_Float16*)take((size_t)G4H * EPAD * 2);
  _Float16* Whh16f  = (_Float16*)take((size_t)G4H * HDIM * 2);
  _Float16* Whh16b  = (_Float16*)take((size_t)G4H * HDIM * 2);
  _Float16* Wout16  = (_Float16*)take((size_t)16 * 256 * 2);
  float*    Gfw     = (float*)take((size_t)BS * G4H * 4);
  float*    Gbw     = (float*)take((size_t)BS * G4H * 4);
  _Float16* hcat    = (_Float16*)take((size_t)BS * 256 * 2);
  float*    Yws     = (float*)take((size_t)BS * 16 * 4);

  // 1) gather + convert embeddings
  k_gather<<<BS, EPAD, 0, stream>>>(x, embed, emb16);
  // 2) weight conversions (zero-padded)
  k_convert<<<(G4H * EPAD + 255) / 256, 256, 0, stream>>>(Wih_f, Wih16f, G4H, 300, G4H, EPAD);
  k_convert<<<(G4H * EPAD + 255) / 256, 256, 0, stream>>>(Wih_b, Wih16b, G4H, 300, G4H, EPAD);
  k_convert<<<(G4H * HDIM + 255) / 256, 256, 0, stream>>>(Whh_f, Whh16f, G4H, HDIM, G4H, HDIM);
  k_convert<<<(G4H * HDIM + 255) / 256, 256, 0, stream>>>(Whh_b, Whh16b, G4H, HDIM, G4H, HDIM);
  k_convert<<<(16 * 256 + 255) / 256, 256, 0, stream>>>(Wout, Wout16, TTAG, 256, 16, 256);
  // 3) input projections (time-independent, fully parallel WMMA GEMMs)
  {
    dim3 grid(BS / 16, G4H / (16 * 8));
    k_inproj<<<grid, 256, 0, stream>>>(emb16, Wih16f, b_f, Gfw);
    k_inproj<<<grid, 256, 0, stream>>>(emb16, Wih16b, b_b, Gbw);
  }
  // 4) recurrent LSTM, both directions concurrent in one grid
  {
    const size_t lds = (size_t)G4H * HDIM * 2 + 2 * 16 * HDIM * 2; // 128KB + 8KB
    k_lstm_rec<<<16, 256, lds, stream>>>(Whh16f, Whh16b, Gfw, Gbw, hcat);
  }
  // 5) emissions
  k_emit<<<BS / (16 * 8), 256, 0, stream>>>(hcat, Wout16, bout, Yws);
  // 6) CRF
  k_crf<<<BAT / 16, 256, 0, stream>>>(Yws, y0, trans, out);
}
